// mini_eMLP_60782377173606
// MI455X (gfx1250) — matrix operand order repaired and checked by reading, not compiled, not run on hardware
//
#include <hip/hip_runtime.h>
#include <hip/hip_bf16.h>
#include <math.h>

// ---------------------------------------------------------------------------
// MI455X (gfx1250) SNN forward.
//   - Hoist input GEMM out of the time scan: pre_h[B*T,H] = X @ Wi2h^T + bi2h
//     (53.7 GFLOP) using 64x64 wave tiles: 16 v_wmma_f32_16x16x32_bf16 per
//     32-K step vs 16 b128 loads (1:1 wmma:vmem issue ratio).
//     Pointers are individually named scalars so clang infers GLOBAL address
//     space (global_load_b128, LOADcnt only) instead of FLAT (which also
//     burns DScnt and the LDS datapath).
//   - Sequential scan: recurrent GEMM hs@Wh2h^T is parallelism-limited
//     (M=128), so split K into 4 chunks (4x the waves) and fold the
//     deterministic partial reduction into the membrane-update kernel.
//   - bf16 A/B with f32 accumulation: 8x the matrix ceiling of the f32
//     16x16x4 WMMA; hs is exactly {0,1} so the recurrent A is lossless.
// ---------------------------------------------------------------------------

typedef __attribute__((ext_vector_type(16))) __bf16 v16bf;
typedef __attribute__((ext_vector_type(8)))  float  v8f;
typedef __attribute__((ext_vector_type(4)))  float  f32x4;

#define B_  128
#define T_  100
#define D_  2048
#define H_  1024
#define O_  20
#define W_  10
#define KC_ 4      // K-split chunks for the recurrent GEMM

union ABFrag { v16bf v; f32x4 f[2]; };

#define LOAD_A(fr, p)  { (fr).f[0] = *(const f32x4*)(p); (fr).f[1] = *(const f32x4*)((p) + 16); }
#define LOAD_B(fr, p)  { (fr).f[0] = *(const f32x4*)(p); (fr).f[1] = *(const f32x4*)((p) + 16); }
#define WMMA(ci, a, b) (ci) = __builtin_amdgcn_wmma_f32_16x16x32_bf16( \
                          false, (a).v, false, (b).v, (short)0, (ci), false, false)

// ----------------------------- conversion ----------------------------------
__global__ __launch_bounds__(256) void f32_to_bf16_kernel(
    const float* __restrict__ src, __bf16* __restrict__ dst, long long n)
{
  long long i = (long long)blockIdx.x * 256 + threadIdx.x;
  if (i < n) dst[i] = (__bf16)src[i];
}

// ----------------------------- state init ----------------------------------
__global__ __launch_bounds__(256) void init_state_kernel(
    const float* __restrict__ hm_in, const float* __restrict__ hs_in,
    const float* __restrict__ om_in,
    float* __restrict__ hm, __bf16* __restrict__ hs_bf,
    float* __restrict__ om, float* __restrict__ osum, float* __restrict__ nbr)
{
  int i = blockIdx.x * 256 + threadIdx.x;
  if (i < B_ * H_) { hm[i] = hm_in[i]; hs_bf[i] = (__bf16)hs_in[i]; }
  if (i < B_ * O_) { om[i] = om_in[i]; osum[i] = 0.f; }
  if (i < T_)      { nbr[i] = 0.f; }
}

// ------------------- big GEMM: 64x64 tile per wave -------------------------
// C[M,N] = A[M,K](bf16) @ Bw[N,K]^T(bf16) + bias[N]
// A-frag (16-bit 16x32): lane L holds row m+(L&15); K-half selected by L>>4.
// B-frag (32x16): lane L holds column n+(L&15) = row n of Bw; K-half selected by L>>4 (same K order as A).
__global__ __launch_bounds__(32) void gemm_bf16_wmma_64x64(
    const __bf16* __restrict__ A, const __bf16* __restrict__ Bw,
    const float* __restrict__ bias, float* __restrict__ C, int N, int K)
{
  const int lane = threadIdx.x;
  const int l16  = lane & 15;
  const int h    = lane >> 4;
  const int m0   = blockIdx.y * 64;
  const int n0   = blockIdx.x * 64;

  v8f c00 = {}, c01 = {}, c02 = {}, c03 = {};
  v8f c10 = {}, c11 = {}, c12 = {}, c13 = {};
  v8f c20 = {}, c21 = {}, c22 = {}, c23 = {};
  v8f c30 = {}, c31 = {}, c32 = {}, c33 = {};

  const __bf16* a0p = A + (size_t)(m0 + l16) * K + h * 8;
  const __bf16* a1p = a0p + (size_t)16 * K;
  const __bf16* a2p = a0p + (size_t)32 * K;
  const __bf16* a3p = a0p + (size_t)48 * K;
  const __bf16* b0p = Bw + (size_t)(n0 + l16) * K + h * 8;
  const __bf16* b1p = b0p + (size_t)16 * K;
  const __bf16* b2p = b0p + (size_t)32 * K;
  const __bf16* b3p = b0p + (size_t)48 * K;

  for (int kk = 0; kk < K; kk += 32) {
    ABFrag af0, af1, af2, af3, bf0, bf1, bf2, bf3;
    LOAD_A(af0, a0p); LOAD_A(af1, a1p); LOAD_A(af2, a2p); LOAD_A(af3, a3p);
    LOAD_B(bf0, b0p); LOAD_B(bf1, b1p); LOAD_B(bf2, b2p); LOAD_B(bf3, b3p);
    a0p += 32; a1p += 32; a2p += 32; a3p += 32;
    b0p += 32; b1p += 32; b2p += 32; b3p += 32;
    if (kk + 32 < K) {                      // global_prefetch_b8 next K tile
      __builtin_prefetch(a0p, 0, 1); __builtin_prefetch(a1p, 0, 1);
      __builtin_prefetch(a2p, 0, 1); __builtin_prefetch(a3p, 0, 1);
      __builtin_prefetch(b0p, 0, 1); __builtin_prefetch(b1p, 0, 1);
      __builtin_prefetch(b2p, 0, 1); __builtin_prefetch(b3p, 0, 1);
    }
    WMMA(c00, af0, bf0); WMMA(c01, af0, bf1); WMMA(c02, af0, bf2); WMMA(c03, af0, bf3);
    WMMA(c10, af1, bf0); WMMA(c11, af1, bf1); WMMA(c12, af1, bf2); WMMA(c13, af1, bf3);
    WMMA(c20, af2, bf0); WMMA(c21, af2, bf1); WMMA(c22, af2, bf2); WMMA(c23, af2, bf3);
    WMMA(c30, af3, bf0); WMMA(c31, af3, bf1); WMMA(c32, af3, bf2); WMMA(c33, af3, bf3);
  }

  // C/D layout: VGPR r, lanes0-15 -> (M=r, N=lane); lanes16-31 -> (M=r+8).
  #define STORE_C(cij, i, j)                                              \
    {                                                                     \
      const int n = n0 + 16 * (j) + l16;                                  \
      const float bv = bias ? bias[n] : 0.f;                              \
      _Pragma("unroll")                                                   \
      for (int r = 0; r < 8; ++r) {                                       \
        const int m = m0 + 16 * (i) + r + 8 * h;                          \
        C[(size_t)m * N + n] = (cij)[r] + bv;                             \
      }                                                                   \
    }
  STORE_C(c00, 0, 0); STORE_C(c01, 0, 1); STORE_C(c02, 0, 2); STORE_C(c03, 0, 3);
  STORE_C(c10, 1, 0); STORE_C(c11, 1, 1); STORE_C(c12, 1, 2); STORE_C(c13, 1, 3);
  STORE_C(c20, 2, 0); STORE_C(c21, 2, 1); STORE_C(c22, 2, 2); STORE_C(c23, 2, 3);
  STORE_C(c30, 3, 0); STORE_C(c31, 3, 1); STORE_C(c32, 3, 2); STORE_C(c33, 3, 3);
  #undef STORE_C
}

// ------------- recurrent GEMM: 32x32 tile, K-split partials ----------------
// part[kc, b, n] = hs[b, koff:koff+256] @ Wh2h[n, koff:koff+256]^T
__global__ __launch_bounds__(32) void gemm_rec_partial(
    const __bf16* __restrict__ A,   // hs_bf [B,H]
    const __bf16* __restrict__ Bw,  // W2bf  [H,H]
    float* __restrict__ part)       // [KC_, B, H]
{
  const int lane = threadIdx.x;
  const int l16  = lane & 15;
  const int h    = lane >> 4;
  const int m0   = blockIdx.y * 32;
  const int n0   = blockIdx.x * 32;
  const int kc   = blockIdx.z;
  const int K    = H_;
  const int klen = H_ / KC_;
  const int koff = kc * klen;

  v8f c00 = {}, c01 = {}, c10 = {}, c11 = {};

  const __bf16* a0p = A  + (size_t)(m0 + l16) * K + koff + h * 8;
  const __bf16* a1p = a0p + (size_t)16 * K;
  const __bf16* b0p = Bw + (size_t)(n0 + l16) * K + koff + h * 8;
  const __bf16* b1p = b0p + (size_t)16 * K;

  for (int kk = 0; kk < klen; kk += 32) {
    ABFrag a0, a1, b0, b1;
    LOAD_A(a0, a0p); LOAD_A(a1, a1p);
    LOAD_B(b0, b0p); LOAD_B(b1, b1p);
    a0p += 32; a1p += 32; b0p += 32; b1p += 32;
    WMMA(c00, a0, b0); WMMA(c01, a0, b1);
    WMMA(c10, a1, b0); WMMA(c11, a1, b1);
  }

  float* Cp = part + (size_t)kc * B_ * H_;
  const int nA = n0 + l16;
  const int nB = nA + 16;
  #pragma unroll
  for (int r = 0; r < 8; ++r) {
    const int mA = m0 + r + 8 * h;
    const int mB = mA + 16;
    Cp[(size_t)mA * H_ + nA] = c00[r];
    Cp[(size_t)mA * H_ + nB] = c01[r];
    Cp[(size_t)mB * H_ + nA] = c10[r];
    Cp[(size_t)mB * H_ + nB] = c11[r];
  }
}

// ------------- hidden mem_update (+ deterministic K-split reduce) ----------
__global__ __launch_bounds__(256) void hid_update_kernel(
    const float* __restrict__ part,    // [KC_, B, H]
    const float* __restrict__ preh_t,  // preh + t*H (row stride T_*H_)
    const float* __restrict__ bh2h,
    float* __restrict__ hm, const float* __restrict__ thr_h,
    __bf16* __restrict__ hs_bf, __bf16* __restrict__ hsT_t,
    float* __restrict__ nbr_t)
{
  __shared__ float bsum;
  const int tid = threadIdx.x;
  if (tid == 0) bsum = 0.f;
  __syncthreads();

  const int i    = blockIdx.x * 256 + tid;
  const int hcol = i & (H_ - 1);
  const int b    = i >> 10;           // H_ == 1024

  float hin = bh2h[hcol] + preh_t[(size_t)b * T_ * H_ + hcol];
  #pragma unroll
  for (int kc = 0; kc < KC_; ++kc) hin += part[(size_t)kc * B_ * H_ + i];

  const float thr = thr_h[hcol];
  float mem = hm[i] + hin;
  const float s = (mem - thr) > 0.f ? 1.f : 0.f;
  mem *= (1.f - s);
  const float neg = (mem < -thr) ? 1.f : 0.f;
  mem = mem * (1.f - neg) - thr * neg;
  hm[i] = mem;
  const __bf16 sb = (__bf16)s;
  hs_bf[i] = sb;   // recurrent operand for next step (exact in bf16)
  hsT_t[i] = sb;   // history for moving-average filter
  if (s != 0.f) atomicAdd(&bsum, 1.f);   // ds_add_f32
  __syncthreads();
  if (tid == 0 && bsum != 0.f) atomicAdd(nbr_t, bsum);
}

// ----------------------------- output layer --------------------------------
__global__ __launch_bounds__(256) void out_layer_kernel(
    const __bf16* __restrict__ hs_bf, const float* __restrict__ Wh2o,
    const float* __restrict__ bh2o, const float* __restrict__ thr_o,
    float* __restrict__ om, float* __restrict__ osum, float* __restrict__ nbr_t)
{
  __shared__ float acc[O_];
  __shared__ float ssum;
  const int b = blockIdx.x, tid = threadIdx.x;
  if (tid < O_) acc[tid] = bh2o[tid];
  if (tid == 0) ssum = 0.f;
  __syncthreads();

  float part[O_];
  #pragma unroll
  for (int o = 0; o < O_; ++o) part[o] = 0.f;
  for (int hh = tid; hh < H_; hh += 256) {
    const float hv = (float)hs_bf[(size_t)b * H_ + hh];
    if (hv != 0.f) {      // spikes are {0,1}: multiply-free gather of weights
      #pragma unroll
      for (int o = 0; o < O_; ++o) part[o] += Wh2o[(size_t)o * H_ + hh];
    }
  }
  #pragma unroll
  for (int o = 0; o < O_; ++o)
    if (part[o] != 0.f) atomicAdd(&acc[o], part[o]);
  __syncthreads();

  if (tid < O_) {
    const int idx = b * O_ + tid;
    const float thr = thr_o[tid];
    float mem = om[idx] + acc[tid];
    const float s = (mem - thr) > 0.f ? 1.f : 0.f;
    mem *= (1.f - s);
    const float neg = (mem < -thr) ? 1.f : 0.f;
    mem = mem * (1.f - neg) - thr * neg;
    om[idx] = mem;
    osum[idx] += s;
    if (s != 0.f) atomicAdd(&ssum, 1.f);
  }
  __syncthreads();
  if (tid == 0 && ssum != 0.f) atomicAdd(nbr_t, ssum);
}

// ------------------------ moving-average filter ----------------------------
__global__ __launch_bounds__(256) void filter_kernel(
    const __bf16* __restrict__ hsT, float* __restrict__ filt_out)
{
  const int i  = blockIdx.x * 256 + threadIdx.x;   // over B*H
  const int BH = B_ * H_;
  float s = 0.f;
  #pragma unroll
  for (int t = 0; t < W_; ++t) s += (float)hsT[(size_t)t * BH + i];
  float* outp = filt_out + (size_t)i * (T_ - W_ + 1);
  outp[0] = s * (1.f / W_);
  for (int j = 1; j <= T_ - W_; ++j) {
    s += (float)hsT[(size_t)(j + W_ - 1) * BH + i]
       - (float)hsT[(size_t)(j - 1) * BH + i];
    outp[j] = s * (1.f / W_);
  }
}

// ---------------------- loss / predictions / nbr ---------------------------
__global__ __launch_bounds__(128) void final_kernel(
    const float* __restrict__ osum, const int* __restrict__ labels,
    const float* __restrict__ nbr, float* __restrict__ out)
{
  __shared__ float lbuf[B_];
  const int b = threadIdx.x;
  float row[O_];
  float mx = -1e30f; int pred = 0;
  #pragma unroll
  for (int o = 0; o < O_; ++o) {
    const float v = osum[b * O_ + o];
    row[o] = v;
    if (v > mx) { mx = v; pred = o; }   // first max == jnp.argmax
  }
  float se = 0.f;
  #pragma unroll
  for (int o = 0; o < O_; ++o) se += __expf(row[o] - mx);
  const float lse = mx + __logf(se);
  lbuf[b] = -(row[labels[b]] - lse);
  out[b] = (float)pred;                 // predictions [1,B]
  __syncthreads();
  if (b == 0) {
    float tot = 0.f;
    for (int k = 0; k < B_; ++k) tot += lbuf[k];
    out[B_] = tot / (float)B_;          // loss
  }
  if (b < T_) out[B_ + 1 + b] = nbr[b] * (1.f / (float)B_);  // nbr_events [T]
}

// ---------------------------------------------------------------------------
extern "C" void kernel_launch(void* const* d_in, const int* in_sizes, int n_in,
                              void* d_out, int out_size, void* d_ws, size_t ws_size,
                              hipStream_t stream)
{
  (void)in_sizes; (void)n_in; (void)out_size; (void)ws_size;
  const float* x      = (const float*)d_in[0];
  const int*   labels = (const int*)  d_in[1];
  const float* hm_in  = (const float*)d_in[2];
  const float* hs_in  = (const float*)d_in[3];
  const float* om_in  = (const float*)d_in[4];
  /* d_in[5] output_spike: dead carry, never read */
  const float* Wi2h  = (const float*)d_in[6];
  const float* bi2h  = (const float*)d_in[7];
  const float* Wh2h  = (const float*)d_in[8];
  const float* bh2h  = (const float*)d_in[9];
  const float* Wh2o  = (const float*)d_in[10];
  const float* bh2o  = (const float*)d_in[11];
  const float* thr_h = (const float*)d_in[12];
  const float* thr_o = (const float*)d_in[13];

  char* ws = (char*)d_ws;
  size_t off = 0;
  auto alloc = [&](size_t bytes) -> char* {
    char* p = ws + off; off += (bytes + 255) & ~(size_t)255; return p;
  };

  __bf16* Xbf  = (__bf16*)alloc((size_t)B_ * T_ * D_ * 2);  // 52.4 MB
  __bf16* hsT  = (__bf16*)Xbf;          // alias: X dead after big GEMM (26.2MB)
  __bf16* W1bf = (__bf16*)alloc((size_t)H_ * D_ * 2);
  __bf16* W2bf = (__bf16*)alloc((size_t)H_ * H_ * 2);
  float*  preh = (float*) alloc((size_t)B_ * T_ * H_ * 4);  // 52.4 MB
  float*  part = (float*) alloc((size_t)KC_ * B_ * H_ * 4); // 2 MB
  __bf16* hsbf = (__bf16*)alloc((size_t)B_ * H_ * 2);
  float*  hm   = (float*) alloc((size_t)B_ * H_ * 4);
  float*  om   = (float*) alloc((size_t)B_ * O_ * 4);
  float*  osum = (float*) alloc((size_t)B_ * O_ * 4);
  float*  nbr  = (float*) alloc((size_t)T_ * 4);

  const long long nX  = (long long)B_ * T_ * D_;
  const long long nW1 = (long long)H_ * D_;
  const long long nW2 = (long long)H_ * H_;
  f32_to_bf16_kernel<<<(nX  + 255) / 256, 256, 0, stream>>>(x,    Xbf,  nX);
  f32_to_bf16_kernel<<<(nW1 + 255) / 256, 256, 0, stream>>>(Wi2h, W1bf, nW1);
  f32_to_bf16_kernel<<<(nW2 + 255) / 256, 256, 0, stream>>>(Wh2h, W2bf, nW2);

  init_state_kernel<<<(B_ * H_ + 255) / 256, 256, 0, stream>>>(
      hm_in, hs_in, om_in, hm, hsbf, om, osum, nbr);

  // Hoisted input GEMM: pre_h[(b*T+t), n] = X @ Wi2h^T + bi2h
  gemm_bf16_wmma_64x64<<<dim3(H_ / 64, (B_ * T_) / 64), 32, 0, stream>>>(
      Xbf, W1bf, bi2h, preh, H_, D_);

  // Sequential scan over T (recurrent dependency)
  for (int t = 0; t < T_; ++t) {
    gemm_rec_partial<<<dim3(H_ / 32, B_ / 32, KC_), 32, 0, stream>>>(
        hsbf, W2bf, part);
    hid_update_kernel<<<(B_ * H_) / 256, 256, 0, stream>>>(
        part, preh + (size_t)t * H_, bh2h, hm, thr_h,
        hsbf, hsT + (size_t)t * B_ * H_, nbr + t);
    out_layer_kernel<<<B_, 256, 0, stream>>>(
        hsbf, Wh2o, bh2o, thr_o, om, osum, nbr + t);
  }

  float* out = (float*)d_out;
  filter_kernel<<<(B_ * H_) / 256, 256, 0, stream>>>(hsT, out + B_ + 1 + T_);
  final_kernel<<<1, B_, 0, stream>>>(osum, labels, nbr, out);
}
